// MambaFusionCore_64209761075742
// MI455X (gfx1250) — matrix operand order repaired and checked by reading, not compiled, not run on hardware
//
#include <hip/hip_runtime.h>
#include <stdint.h>

// ---------------- problem constants ----------------
constexpr int kB  = 4;
constexpr int kL  = 2048;
constexpr int kH  = 1024;
constexpr int kNL = 4;
constexpr int kDS = 16;
constexpr int kDC = 4;
constexpr int kDI = 2048;          // H * EX
constexpr int kM  = kB * kL;       // 8192 rows

typedef uint16_t u16;
typedef __bf16 bf16_t;
typedef bf16_t v16bf __attribute__((ext_vector_type(16)));
typedef bf16_t bf8v  __attribute__((ext_vector_type(8)));
typedef float  v8f   __attribute__((ext_vector_type(8)));

// ---------------- small helpers ----------------
static __device__ __forceinline__ u16 f2bf_bits(float f) {
    uint32_t u = __builtin_bit_cast(uint32_t, f);
    uint32_t r = (u + 0x7FFFu + ((u >> 16) & 1u)) >> 16;   // RNE
    return (u16)r;
}
static __device__ __forceinline__ float bf2f(u16 h) {
    uint32_t u = ((uint32_t)h) << 16;
    return __builtin_bit_cast(float, u);
}
static __device__ __forceinline__ float siluf(float v) {
    return v / (1.0f + __expf(-v));
}
static __device__ __forceinline__ float sigmoidf_(float v) {
    return 1.0f / (1.0f + __expf(-v));
}
static __device__ __forceinline__ float softplusf_(float v) {
    return (v > 20.0f) ? v : log1pf(__expf(v));
}

// -------- CDNA5 async copy to LDS (ASYNCcnt-tracked DMA, no VGPR round-trip) --------
static __device__ __forceinline__ void async_ld_b128(uint32_t lds_off, const void* gptr) {
    asm volatile("global_load_async_to_lds_b128 %0, %1, off"
                 :: "v"(lds_off), "v"((uint64_t)(uintptr_t)gptr)
                 : "memory");
}
static __device__ __forceinline__ void wait_async0() {
    asm volatile("s_wait_asynccnt 0x0" ::: "memory");
}

// ---------------- f32 -> bf16 convert ----------------
__global__ void k_f32_to_bf16(const float* __restrict__ in, u16* __restrict__ out, int n) {
    int i = blockIdx.x * blockDim.x + threadIdx.x;
    if (i < n) out[i] = f2bf_bits(in[i]);
}

// ---------------- row LayerNorm (one block / row) ----------------
__global__ void k_ln_rows(const float* __restrict__ x, const float* __restrict__ w,
                          const float* __restrict__ b, float* __restrict__ outF,
                          u16* __restrict__ outBf, int W) {
    const int row = blockIdx.x;
    const float* xr = x + (size_t)row * W;
    __shared__ float s1[256], s2[256];
    float sum = 0.f, sq = 0.f;
    for (int i = threadIdx.x; i < W; i += 256) { float v = xr[i]; sum += v; sq += v * v; }
    s1[threadIdx.x] = sum; s2[threadIdx.x] = sq;
    __syncthreads();
    for (int s = 128; s > 0; s >>= 1) {
        if (threadIdx.x < s) { s1[threadIdx.x] += s1[threadIdx.x + s]; s2[threadIdx.x] += s2[threadIdx.x + s]; }
        __syncthreads();
    }
    const float mean = s1[0] / W;
    const float var  = s2[0] / W - mean * mean;
    const float rstd = rsqrtf(var + 1e-5f);
    for (int i = threadIdx.x; i < W; i += 256) {
        float v = (xr[i] - mean) * rstd * w[i] + b[i];
        if (outF)  outF[(size_t)row * W + i] = v;
        if (outBf) outBf[(size_t)row * W + i] = f2bf_bits(v);
    }
}

// ---------------- WMMA fragment loaders (contiguous K order, A and B alike) ----------------
// A (16x32 bf16): lane half h holds 16 contiguous K at k0+16h
static __device__ __forceinline__ v16bf load_fragA(const bf16_t* __restrict__ base,
                                                   int row, int K, int k0, int half) {
    const bf16_t* p = base + (size_t)row * K + k0;
    bf8v lo = *(const bf8v*)(p + half * 16);
    bf8v hi = *(const bf8v*)(p + half * 16 + 8);
    v16bf f;
#pragma unroll
    for (int i = 0; i < 8; ++i) { f[i] = lo[i]; f[i + 8] = hi[i]; }
    return f;
}
// B (32x16 bf16, fed with row-major W[n][k]): lane half h holds 16 contiguous K at k0+16h
static __device__ __forceinline__ v16bf load_fragB(const bf16_t* __restrict__ base,
                                                   int row, int K, int k0, int half) {
    const bf16_t* p = base + (size_t)row * K + k0 + half * 16;
    bf8v lo = *(const bf8v*)(p);
    bf8v hi = *(const bf8v*)(p + 8);
    v16bf f;
#pragma unroll
    for (int i = 0; i < 8; ++i) { f[i] = lo[i]; f[i + 8] = hi[i]; }
    return f;
}
// B fragment from an LDS tile: row-major 128 rows x 32 K (u16), row stride 32 elems
static __device__ __forceinline__ v16bf load_fragB_lds(const u16* __restrict__ tile,
                                                       int row, int half) {
    const bf16_t* p = (const bf16_t*)(tile + row * 32 + half * 16);
    bf8v lo = *(const bf8v*)(p);
    bf8v hi = *(const bf8v*)(p + 8);
    v16bf f;
#pragma unroll
    for (int i = 0; i < 8; ++i) { f[i] = lo[i]; f[i + 8] = hi[i]; }
    return f;
}

// ---------------- fused epilogues ----------------
enum { EPI_F32 = 0, EPI_SPLIT_SILU = 1, EPI_ADDRES = 2, EPI_GELU_BF = 3, EPI_MLP2 = 4 };

static __device__ __forceinline__ void epilogue_store(
        int m, int n, int N, float v,
        float* __restrict__ outF, u16* __restrict__ outBfA, u16* __restrict__ outBfB,
        const float* __restrict__ bias, const float* __restrict__ resF,
        int mode, int halfN) {
    if (bias) v += bias[n];
    const size_t idx = (size_t)m * N + n;
    switch (mode) {
        case EPI_F32:
            outF[idx] = v; break;
        case EPI_SPLIT_SILU:
            if (n < halfN) outBfA[(size_t)m * halfN + n] = f2bf_bits(siluf(v));
            else           outBfB[(size_t)m * halfN + (n - halfN)] = f2bf_bits(v);
            break;
        case EPI_ADDRES:
            outF[idx] = resF[idx] + v; break;
        case EPI_GELU_BF: {
            float g = 0.5f * v * (1.0f + erff(v * 0.70710678118654752f));
            outBfA[idx] = f2bf_bits(g);
        } break;
        case EPI_MLP2:
            outF[idx] = v + resF[idx]; break;
    }
}

// ---------------- big WMMA GEMM (N % 128 == 0, M % 128 == 0) ----------------
// Block: 256 thr = 8 waves; block tile 128(M) x 128(N).
// Wave tile 32(M) x 64(N): 2 A-frags x 4 B-frags -> 8 WMMA per k-step.
// B tile (128 x 32 bf16 = 8KB) staged to LDS by async-DMA, double-buffered so the
// DMA for step k+1 overlaps the 8 WMMAs of step k (ASYNCcnt + workgroup barrier).
__global__ void k_gemm_wmma_big(const u16* __restrict__ Abf, const u16* __restrict__ Wbf,
                                int M, int N, int K,
                                float* __restrict__ outF, u16* __restrict__ outBfA,
                                u16* __restrict__ outBfB, const float* __restrict__ bias,
                                const float* __restrict__ resF, int mode, int halfN) {
    __shared__ u16 sB[2][128 * 32];          // 16 KB double buffer

    const int tid   = threadIdx.x;
    const int wave  = tid >> 5;
    const int lane  = tid & 31;
    const int half  = lane >> 4;
    const int r     = lane & 15;
    const int waveM = wave & 3;
    const int waveN = wave >> 2;
    const int mTile = blockIdx.y * 128 + waveM * 32;
    const int nBlk  = blockIdx.x * 128;
    const bf16_t* A = (const bf16_t*)Abf;

    // stage B tile rows [nBlk, nBlk+128), K cols [k0, k0+32) into sB[buf]
    auto stage = [&](int k0, int buf) {
#pragma unroll
        for (int t = 0; t < 2; ++t) {
            const int i     = tid + t * 256;      // 0..511 -> 512 x 16B = 8KB
            const int row   = i >> 2;
            const int chunk = i & 3;
            const u16* g = Wbf + (size_t)(nBlk + row) * K + k0 + chunk * 8;
            const uint32_t lds = (uint32_t)(uintptr_t)(&sB[buf][row * 32 + chunk * 8]);
            async_ld_b128(lds, g);
        }
    };

    stage(0, 0);

    v8f acc[2][4] = {};
    const int aRow0 = mTile + r;
    const int aRow1 = mTile + 16 + r;
    int buf = 0;

    for (int k0 = 0; k0 < K; k0 += 32) {
        wait_async0();           // my DMAs for sB[buf] done
        __syncthreads();         // everyone's DMAs for sB[buf] done
        if (k0 + 32 < K) stage(k0 + 32, buf ^ 1);   // overlap next tile's DMA
        if (k0 + 64 < K) {
            __builtin_prefetch(A + (size_t)aRow0 * K + k0 + 64, 0, 1);
            __builtin_prefetch(A + (size_t)aRow1 * K + k0 + 64, 0, 1);
        }
        v16bf a0 = load_fragA(A, aRow0, K, k0, half);
        v16bf a1 = load_fragA(A, aRow1, K, k0, half);
        const u16* tile = sB[buf];
#pragma unroll
        for (int f = 0; f < 4; ++f) {
            v16bf bf = load_fragB_lds(tile, waveN * 64 + f * 16 + r, half);
            acc[0][f] = __builtin_amdgcn_wmma_f32_16x16x32_bf16(
                false, a0, false, bf, (short)0, acc[0][f], false, false);
            acc[1][f] = __builtin_amdgcn_wmma_f32_16x16x32_bf16(
                false, a1, false, bf, (short)0, acc[1][f], false, false);
        }
        buf ^= 1;
    }

    const int nTile = nBlk + waveN * 64;
#pragma unroll
    for (int mi = 0; mi < 2; ++mi)
#pragma unroll
        for (int f = 0; f < 4; ++f) {
            const int n = nTile + f * 16 + r;
#pragma unroll
            for (int i = 0; i < 8; ++i) {
                const int m = mTile + mi * 16 + i + 8 * half;
                epilogue_store(m, n, N, acc[mi][f][i], outF, outBfA, outBfB,
                               bias, resF, mode, halfN);
            }
        }
}

// ---------------- small (column-guarded) WMMA GEMM, for N=33 x_proj ----------------
// Block: 256 thr = 8 waves stacked along M; wave tile 16 x 64.
__global__ void k_gemm_wmma_small(const u16* __restrict__ Abf, const u16* __restrict__ Wbf,
                                  int M, int N, int K, float* __restrict__ outF) {
    const int wave = threadIdx.x >> 5;
    const int lane = threadIdx.x & 31;
    const int half = lane >> 4;
    const int r    = lane & 15;
    const int mTile = blockIdx.y * 128 + wave * 16;
    const int nTile = blockIdx.x * 64;
    const bf16_t* A = (const bf16_t*)Abf;
    const bf16_t* W = (const bf16_t*)Wbf;

    v8f acc[4] = {};
    const int aRow = mTile + r;

    for (int k0 = 0; k0 < K; k0 += 32) {
        v16bf af = load_fragA(A, aRow, K, k0, half);
#pragma unroll
        for (int f = 0; f < 4; ++f) {
            const int n = nTile + f * 16 + r;
            v16bf bf;
            if (n < N) bf = load_fragB(W, n, K, k0, half);
            else       bf = v16bf{};
            acc[f] = __builtin_amdgcn_wmma_f32_16x16x32_bf16(
                false, af, false, bf, (short)0, acc[f], false, false);
        }
    }

#pragma unroll
    for (int f = 0; f < 4; ++f) {
        const int n = nTile + f * 16 + r;
        if (n >= N) continue;
#pragma unroll
        for (int i = 0; i < 8; ++i) {
            const int m = mTile + i + 8 * half;
            outF[(size_t)m * N + n] = acc[f][i];
        }
    }
}

// ---------------- depthwise causal conv (DC=4) + bias + SiLU ----------------
__global__ void k_dwconv_silu(const u16* __restrict__ in, const float* __restrict__ w,
                              const float* __restrict__ bias, u16* __restrict__ out) {
    const int idx = blockIdx.x * blockDim.x + threadIdx.x;   // over M*DI
    if (idx >= kM * kDI) return;
    const int d  = idx % kDI;
    const int ml = idx / kDI;
    const int l  = ml % kL;
    const int bb = ml / kL;
    float acc = bias[d];
#pragma unroll
    for (int j = 0; j < kDC; ++j) {
        const int ls = l - (kDC - 1) + j;
        if (ls >= 0)
            acc += w[d * kDC + j] * bf2f(in[((size_t)(bb * kL + ls)) * kDI + d]);
    }
    out[idx] = f2bf_bits(siluf(acc));
}

// ---------------- selective scan: one thread per (b, d) channel ----------------
__global__ void k_scan(const float* __restrict__ xp,   // (M, 33): [dt_r | B(16) | C(16)]
                       const u16* __restrict__ xin,    // (M, DI) bf16 post-conv
                       const float* __restrict__ dtw,  // (DI)
                       const float* __restrict__ dtb,  // (DI)
                       const float* __restrict__ Dp,   // (DI)
                       float* __restrict__ y) {        // (M, DI)
    const int chunk = blockIdx.x & ((kDI / 256) - 1);
    const int bb    = blockIdx.x / (kDI / 256);
    const int d     = chunk * 256 + threadIdx.x;
    const float dtwd = dtw[d], dtbd = dtb[d], Dd = Dp[d];
    float h[kDS];
#pragma unroll
    for (int n = 0; n < kDS; ++n) h[n] = 0.f;
    for (int l = 0; l < kL; ++l) {
        const size_t base = (size_t)(bb * kL + l);
        const float* row = xp + base * 33;
        const float dtr = row[0];
        const float xt  = bf2f(xin[base * kDI + d]);
        const float dt  = softplusf_(dtr * dtwd + dtbd);
        float acc = 0.f;
#pragma unroll
        for (int n = 0; n < kDS; ++n) {
            const float Bn = row[1 + n];
            const float Cn = row[1 + kDS + n];
            const float An = -(float)(n + 1);
            h[n] = h[n] + dt * (h[n] * An + xt * Bn);
            acc += h[n] * Cn;
        }
        y[base * kDI + d] = acc + xt * Dd;
    }
}

// ---------------- group-norm over DI + sigmoid gate ----------------
__global__ void k_gate(const float* __restrict__ y, const u16* __restrict__ resBf,
                       const float* __restrict__ mnw, const float* __restrict__ mnb,
                       u16* __restrict__ outBf) {
    const int row = blockIdx.x;
    const float* yr = y + (size_t)row * kDI;
    __shared__ float s1[256], s2[256];
    float sum = 0.f, sq = 0.f;
    for (int i = threadIdx.x; i < kDI; i += 256) { float v = yr[i]; sum += v; sq += v * v; }
    s1[threadIdx.x] = sum; s2[threadIdx.x] = sq;
    __syncthreads();
    for (int s = 128; s > 0; s >>= 1) {
        if (threadIdx.x < s) { s1[threadIdx.x] += s1[threadIdx.x + s]; s2[threadIdx.x] += s2[threadIdx.x + s]; }
        __syncthreads();
    }
    const float mean = s1[0] / kDI;
    const float var  = s2[0] / kDI - mean * mean;
    const float rstd = rsqrtf(var + 1e-5f);
    for (int i = threadIdx.x; i < kDI; i += 256) {
        float v = (yr[i] - mean) * rstd * mnw[i] + mnb[i];
        v *= sigmoidf_(bf2f(resBf[(size_t)row * kDI + i]));
        outBf[(size_t)row * kDI + i] = f2bf_bits(v);
    }
}

// ---------------- host side ----------------
static inline void* ws_alloc(char* base, size_t& off, size_t bytes) {
    size_t o = (off + 255) & ~(size_t)255;
    off = o + bytes;
    return (void*)(base + o);
}

extern "C" void kernel_launch(void* const* d_in, const int* in_sizes, int n_in,
                              void* d_out, int out_size, void* d_ws, size_t ws_size,
                              hipStream_t stream) {
    (void)in_sizes; (void)n_in; (void)out_size; (void)ws_size;
    const float* hidden   = (const float*)d_in[0];
    const float* in_nw    = (const float*)d_in[1];
    const float* in_nb    = (const float*)d_in[2];
    const float* ln_w     = (const float*)d_in[3];
    const float* ln_b     = (const float*)d_in[4];
    const float* in_projw = (const float*)d_in[5];
    const float* conv_w   = (const float*)d_in[6];
    const float* conv_b   = (const float*)d_in[7];
    const float* x_projw  = (const float*)d_in[8];
    const float* dt_w     = (const float*)d_in[9];
    const float* dt_b     = (const float*)d_in[10];
    const float* Dp       = (const float*)d_in[11];
    const float* mn_w     = (const float*)d_in[12];
    const float* mn_b     = (const float*)d_in[13];
    const float* out_w    = (const float*)d_in[14];
    const float* op_w1    = (const float*)d_in[15];
    const float* op_b1    = (const float*)d_in[16];
    const float* op_w2    = (const float*)d_in[17];
    const float* op_b2    = (const float*)d_in[18];
    float* out = (float*)d_out;

    char* ws = (char*)d_ws;
    size_t off = 0;
    float* cur_x  = (float*)ws_alloc(ws, off, (size_t)kM * kH * 4);
    float* ybuf   = (float*)ws_alloc(ws, off, (size_t)kM * kDI * 4);
    float* xp     = (float*)ws_alloc(ws, off, (size_t)kM * 33 * 4);
    u16*   xn_bf  = (u16*)  ws_alloc(ws, off, (size_t)kM * kH * 2);
    u16*   xi_bf  = (u16*)  ws_alloc(ws, off, (size_t)kM * kDI * 2);
    u16*   xi2_bf = (u16*)  ws_alloc(ws, off, (size_t)kM * kDI * 2);
    u16*   res_bf = (u16*)  ws_alloc(ws, off, (size_t)kM * kDI * 2);
    u16*   y_bf   = (u16*)  ws_alloc(ws, off, (size_t)kM * kDI * 2);
    u16*   w_bf   = (u16*)  ws_alloc(ws, off, (size_t)(2 * kDI) * kH * 2);  // max weight

    const dim3 blk(256);
    auto cdiv = [](int a, int b) { return (a + b - 1) / b; };

    // x = LN(hidden, in_norm) -> cur_x (f32)
    k_ln_rows<<<kM, blk, 0, stream>>>(hidden, in_nw, in_nb, cur_x, nullptr, kH);

    for (int l = 0; l < kNL; ++l) {
        // xn = LN(cur_x, ln_w[l]) -> bf16
        k_ln_rows<<<kM, blk, 0, stream>>>(cur_x, ln_w + l * kH, ln_b + l * kH,
                                          nullptr, xn_bf, kH);
        // in_proj: (M,H) @ (2DI,H)^T, split + silu
        {
            const int N = 2 * kDI, K = kH;
            const int nelt = N * K;
            k_f32_to_bf16<<<cdiv(nelt, 256), blk, 0, stream>>>(
                in_projw + (size_t)l * nelt, w_bf, nelt);
            dim3 grid(N / 128, kM / 128);
            k_gemm_wmma_big<<<grid, blk, 0, stream>>>(xn_bf, w_bf, kM, N, K,
                nullptr, xi_bf, res_bf, nullptr, nullptr, EPI_SPLIT_SILU, kDI);
        }
        // two depthwise causal convs + silu
        k_dwconv_silu<<<cdiv(kM * kDI, 256), blk, 0, stream>>>(
            xi_bf, conv_w + (size_t)(l * 2 + 0) * kDI * kDC,
            conv_b + (size_t)(l * 2 + 0) * kDI, xi2_bf);
        k_dwconv_silu<<<cdiv(kM * kDI, 256), blk, 0, stream>>>(
            xi2_bf, conv_w + (size_t)(l * 2 + 1) * kDI * kDC,
            conv_b + (size_t)(l * 2 + 1) * kDI, xi_bf);
        // x_proj: (M,DI) @ (33,DI)^T -> xp f32 (guarded small GEMM)
        {
            const int N = 1 + 2 * kDS, K = kDI;
            const int nelt = N * K;
            k_f32_to_bf16<<<cdiv(nelt, 256), blk, 0, stream>>>(
                x_projw + (size_t)l * nelt, w_bf, nelt);
            dim3 grid(cdiv(N, 64), kM / 128);
            k_gemm_wmma_small<<<grid, blk, 0, stream>>>(xi_bf, w_bf, kM, N, K, xp);
        }
        // selective scan
        k_scan<<<kB * (kDI / 256), blk, 0, stream>>>(
            xp, xi_bf, dt_w + (size_t)l * kDI, dt_b + (size_t)l * kDI,
            Dp + (size_t)l * kDI, ybuf);
        // group norm + sigmoid gate -> y_bf
        k_gate<<<kM, blk, 0, stream>>>(ybuf, res_bf, mn_w + (size_t)l * kDI,
                                       mn_b + (size_t)l * kDI, y_bf);
        // out proj + residual: cur_x = cur_x + y @ out_w^T
        {
            const int N = kH, K = kDI;
            const int nelt = N * K;
            k_f32_to_bf16<<<cdiv(nelt, 256), blk, 0, stream>>>(
                out_w + (size_t)l * nelt, w_bf, nelt);
            dim3 grid(N / 128, kM / 128);
            k_gemm_wmma_big<<<grid, blk, 0, stream>>>(y_bf, w_bf, kM, N, K,
                cur_x, nullptr, nullptr, nullptr, cur_x, EPI_ADDRES, 0);
        }
    }

    // MLP: o1 = gelu(cur_x @ w1^T + b1) -> bf16 (into y_bf, width H)
    {
        const int N = kH, K = kH, nelt = N * K;
        k_f32_to_bf16<<<cdiv(kM * kH, 256), blk, 0, stream>>>(cur_x, xn_bf, kM * kH);
        k_f32_to_bf16<<<cdiv(nelt, 256), blk, 0, stream>>>(op_w1, w_bf, nelt);
        dim3 grid(N / 128, kM / 128);
        k_gemm_wmma_big<<<grid, blk, 0, stream>>>(xn_bf, w_bf, kM, N, K,
            nullptr, y_bf, nullptr, op_b1, nullptr, EPI_GELU_BF, 0);
    }
    // o2 = o1 @ w2^T + b2 + hidden -> d_out
    {
        const int N = kH, K = kH, nelt = N * K;
        k_f32_to_bf16<<<cdiv(nelt, 256), blk, 0, stream>>>(op_w2, w_bf, nelt);
        dim3 grid(N / 128, kM / 128);
        k_gemm_wmma_big<<<grid, blk, 0, stream>>>(y_bf, w_bf, kM, N, K,
            out, nullptr, nullptr, op_b2, hidden, EPI_MLP2, 0);
    }
}
